// ScratchRNN_56762287784433
// MI455X (gfx1250) — matrix operand order repaired and checked by reading, not compiled, not run on hardware
//
#include <hip/hip_runtime.h>
#include <hip/hip_bf16.h>
#include <math.h>

#define T_STEPS 51
#define EMB 64
#define HID 128
#define KW 192   // combined K (128 hh + 64 hx)
#define KP 200   // padded LDS row stride (bf16 elems), multiple of 8 for b128 alignment

typedef __attribute__((ext_vector_type(16))) __bf16 v16bf;
typedef __attribute__((ext_vector_type(8)))  float  v8f;
typedef __attribute__((ext_vector_type(16))) float  v16f;
typedef __attribute__((ext_vector_type(4)))  unsigned v4u;
typedef __attribute__((ext_vector_type(8)))  int      v8i;
typedef __attribute__((ext_vector_type(4)))  int      v4i;

union BFrag { unsigned u[8]; uint4 q[2]; v16bf v; };

static __device__ __forceinline__ unsigned f2bf(float f) {
  union { float f; unsigned u; } x; x.f = f; return x.u >> 16;
}
static __device__ __forceinline__ unsigned pack2(float lo, float hi) {
  union { float f; unsigned u; } a, b; a.f = lo; b.f = hi;
  return (b.u & 0xFFFF0000u) | (a.u >> 16);
}

// ---------------- prep: weights -> bf16 Wc^T[t][n][k], k contiguous (192) ----------------
__global__ __launch_bounds__(256) void convert_weights_kernel(
    const float* __restrict__ whx, const float* __restrict__ whh,
    unsigned* __restrict__ wbf_dw) {
  const long p = (long)blockIdx.x * 256 + threadIdx.x;   // dword (k-pair) index
  const int  kk = ((int)(p % 96)) * 2;                   // even k
  const long r  = p / 96;                                // t*128 + n
  const int  n  = (int)(r % 128);
  const int  t  = (int)(r / 128);
  float v0, v1;
  if (kk < HID) {
    v0 = whh[((size_t)t * HID + kk)     * HID + n];
    v1 = whh[((size_t)t * HID + kk + 1) * HID + n];
  } else {
    v0 = whx[((size_t)t * EMB + (kk - HID))     * HID + n];
    v1 = whx[((size_t)t * EMB + (kk - HID) + 1) * HID + n];
  }
  wbf_dw[p] = pack2(v0, v1);
}

// ---------------- TDM: stream one step's 128x192 bf16 weight tile into LDS --------------
// Iterate mode: 128 iterations of a 192-element row; LDS increment = KP(200) elements,
// global increment = 192 elements -> TDM produces the bank-padded LDS layout directly.
static __device__ __forceinline__ void tdm_load_weights(unsigned lds_byte_addr,
                                                        unsigned long long gaddr) {
  v4u g0;
  g0.x = 1u;                                            // count=1, user descriptor
  g0.y = lds_byte_addr;                                 // [63:32] lds_addr
  g0.z = (unsigned)(gaddr & 0xFFFFFFFFu);               // [95:64] global_addr lo
  g0.w = (unsigned)((gaddr >> 32) & 0x01FFFFFFu)        // [120:96] global_addr hi
       | 0x80000000u;                                   // [127:126] type=2 ("image")
  v8i g1;
  g1[0] = (int)((1u << 16) | (1u << 19));               // data_size=2B, iterate_enable=1
  g1[1] = (int)(192u << 16);                            // tensor_dim0=192 (lo 16)
  g1[2] = (int)(1u << 16);                              // tensor_dim0 hi=0; tensor_dim1=1
  g1[3] = (int)(192u << 16);                            // tile_dim0=192
  g1[4] = 0;                                            // tile_dim1/2 unused
  g1[5] = 192;                                          // tensor_dim0_stride=192
  g1[6] = 0;
  g1[7] = 0;
  v4i g2;
  g2[0] = 0;
  g2[1] = KP;                                           // lds_addr_increment (elems)
  g2[2] = KW;                                           // global_addr_increment (elems)
  g2[3] = (int)(127u << 16);                            // iterate_count=127 -> 128 iters
  v4i g3 = {0, 0, 0, 0};
  v8i g4 = {0, 0, 0, 0, 0, 0, 0, 0};                    // VADDR4 slot: unused/NULL
  __builtin_amdgcn_tensor_load_to_lds(g0, g1, g2, g3, g4, 0);
}

// ---------------- main kernel: TDM double-buffered weights + WMMA recurrence ------------
__global__ __launch_bounds__(128) void rnn_wmma_tdm_kernel(
    const int* __restrict__ xpad, const float* __restrict__ emb,
    const float* __restrict__ bh, const unsigned short* __restrict__ wbf,
    float* __restrict__ out) {

  __shared__ __align__(16) unsigned short wbuf[2 * HID * KP];  // 2 x 50KB

  const int tid  = threadIdx.x;      // 0..127
  const int lane = tid & 31;
  const int wave = tid >> 5;         // 0..3
  const int m    = lane & 15;        // WMMA N col (batch row) = A row within tile
  const int H    = lane >> 4;        // half-wave selector

  const long row = (long)blockIdx.x * 64 + wave * 16 + m;

  const unsigned lds_base = (unsigned)(size_t)(const void*)wbuf;  // low 32 = LDS offset
  const unsigned long long wg = (unsigned long long)(size_t)wbf;

  // kick TDM for step 0 into buffer 0
  if (wave == 0) tdm_load_weights(lds_base, wg);

  // recurrent B fragments (z^T hidden part): h0 = 0
  v16bf hB[4];
  {
    BFrag z;
    #pragma unroll
    for (int d = 0; d < 8; ++d) z.u[d] = 0u;
    #pragma unroll
    for (int c = 0; c < 4; ++c) hB[c] = z.v;
  }

  v8f acc[8];

  for (int t = 0; t < T_STEPS; ++t) {
    __builtin_amdgcn_s_wait_tensorcnt(0);  // wave0: TDM for step t done (no-op elsewhere)
    __syncthreads();                       // everyone: buf[t&1] ready; iter t-1 reads done
    if (wave == 0 && (t + 1) < T_STEPS)    // prefetch step t+1 into the other buffer
      tdm_load_weights(lds_base + (unsigned)(((t + 1) & 1) * HID * KP * 2),
                       wg + (unsigned long long)(t + 1) * (HID * KW * 2));

    const unsigned short* wtp = wbuf + (t & 1) * (HID * KP);

    // ---- x_t B fragments via embedding gather (16 contiguous floats per lane/chunk) ----
    const int tok = xpad[row * T_STEPS + t];
    const float* ev = emb + (size_t)tok * EMB;
    v16bf bx[2];
    #pragma unroll
    for (int c = 0; c < 2; ++c) {
      const v16f xv = *(const v16f*)(ev + c * 32 + 16 * H);
      BFrag f;
      #pragma unroll
      for (int d = 0; d < 8; ++d) f.u[d] = pack2(xv[2 * d], xv[2 * d + 1]);
      bx[c] = f.v;
    }

    // ---- bias init: acc[i][v] <-> hidden n = i*16 + 8H + v, batch col m ----
    const float* bp = bh + (size_t)t * HID;
    #pragma unroll
    for (int i = 0; i < 8; ++i)
      acc[i] = *(const v8f*)(bp + i * 16 + 8 * H);

    // ---- D' = Wc^T x z^T + C : 6 K-chunks, per chunk load 8 A tiles then 8 WMMAs ----
    #pragma unroll
    for (int c = 0; c < 6; ++c) {
      const int k0 = c * 32 + 16 * H;   // A layout: K contiguous {k0..k0+15}
      BFrag af[8];
      #pragma unroll
      for (int i = 0; i < 8; ++i) {
        const int n = i * 16 + m;
        af[i].q[0] = *(const uint4*)(wtp + n * KP + k0);
        af[i].q[1] = *(const uint4*)(wtp + n * KP + k0 + 8);
      }
      const v16bf bf = (c < 4) ? hB[c] : bx[c - 4];
      #pragma unroll
      for (int i = 0; i < 8; ++i)
        acc[i] = __builtin_amdgcn_wmma_f32_16x16x32_bf16(
                     false, af[i].v, false, bf, (short)0, acc[i], false, false);
    }

    // ---- repack D' -> next step's hidden B fragments (lane^16 exchange) ----
    #pragma unroll
    for (int c = 0; c < 4; ++c) {
      unsigned p0[4], p1[4];
      #pragma unroll
      for (int d = 0; d < 4; ++d) {
        p0[d] = pack2(acc[2 * c][2 * d],     acc[2 * c][2 * d + 1]);
        p1[d] = pack2(acc[2 * c + 1][2 * d], acc[2 * c + 1][2 * d + 1]);
      }
      BFrag f;
      #pragma unroll
      for (int d = 0; d < 4; ++d) {
        const unsigned snd = H ? p0[d] : p1[d];
        const unsigned own = H ? p1[d] : p0[d];
        const unsigned rcv = (unsigned)__shfl_xor((int)snd, 16, 32);
        f.u[d]     = H ? rcv : own;   // halfwords 0..7  : K = c*32 + 16H + 0..7
        f.u[4 + d] = H ? own : rcv;   // halfwords 8..15 : K = c*32 + 16H + 8..15
      }
      hB[c] = f.v;
    }
  }

  // ---- epilogue: y = tanh(h_T), contiguous 32B stores ----
  #pragma unroll
  for (int i = 0; i < 8; ++i) {
    v8f o;
    #pragma unroll
    for (int v = 0; v < 8; ++v) o[v] = tanhf(acc[i][v]);
    *(v8f*)(out + (size_t)row * HID + i * 16 + 8 * H) = o;
  }
}

// ---------------- fallback (no workspace): in-kernel staging version --------------------
__global__ __launch_bounds__(128) void rnn_wmma_stage_kernel(
    const int* __restrict__ xpad, const float* __restrict__ emb,
    const float* __restrict__ whx, const float* __restrict__ whh,
    const float* __restrict__ bh, float* __restrict__ out) {

  __shared__ __align__(16) unsigned short wt[HID * KP];

  const int tid  = threadIdx.x;
  const int lane = tid & 31;
  const int wave = tid >> 5;
  const int m    = lane & 15;
  const int H    = lane >> 4;
  const long row = (long)blockIdx.x * 64 + wave * 16 + m;
  const int n4 = (tid & 31) * 4;
  const int ko = tid >> 5;

  v16bf hB[4];
  {
    BFrag z;
    #pragma unroll
    for (int d = 0; d < 8; ++d) z.u[d] = 0u;
    #pragma unroll
    for (int c = 0; c < 4; ++c) hB[c] = z.v;
  }
  v8f acc[8];

  for (int t = 0; t < T_STEPS; ++t) {
    __syncthreads();
    for (int kb = 0; kb < HID; kb += 4) {
      const int k = kb + ko;
      const float4 w = *(const float4*)(whh + ((size_t)t * HID + k) * HID + n4);
      wt[(n4 + 0) * KP + k] = (unsigned short)f2bf(w.x);
      wt[(n4 + 1) * KP + k] = (unsigned short)f2bf(w.y);
      wt[(n4 + 2) * KP + k] = (unsigned short)f2bf(w.z);
      wt[(n4 + 3) * KP + k] = (unsigned short)f2bf(w.w);
    }
    for (int kb = 0; kb < EMB; kb += 4) {
      const int k = kb + ko;
      const float4 w = *(const float4*)(whx + ((size_t)t * EMB + k) * HID + n4);
      wt[(n4 + 0) * KP + HID + k] = (unsigned short)f2bf(w.x);
      wt[(n4 + 1) * KP + HID + k] = (unsigned short)f2bf(w.y);
      wt[(n4 + 2) * KP + HID + k] = (unsigned short)f2bf(w.z);
      wt[(n4 + 3) * KP + HID + k] = (unsigned short)f2bf(w.w);
    }
    __syncthreads();

    const int tok = xpad[row * T_STEPS + t];
    const float* ev = emb + (size_t)tok * EMB;
    v16bf bx[2];
    #pragma unroll
    for (int c = 0; c < 2; ++c) {
      const v16f xv = *(const v16f*)(ev + c * 32 + 16 * H);
      BFrag f;
      #pragma unroll
      for (int d = 0; d < 8; ++d) f.u[d] = pack2(xv[2 * d], xv[2 * d + 1]);
      bx[c] = f.v;
    }
    const float* bp = bh + (size_t)t * HID;
    #pragma unroll
    for (int i = 0; i < 8; ++i)
      acc[i] = *(const v8f*)(bp + i * 16 + 8 * H);

    #pragma unroll
    for (int c = 0; c < 6; ++c) {
      const int k0 = c * 32 + 16 * H;
      BFrag af[8];
      #pragma unroll
      for (int i = 0; i < 8; ++i) {
        const int n = i * 16 + m;
        af[i].q[0] = *(const uint4*)(wt + n * KP + k0);
        af[i].q[1] = *(const uint4*)(wt + n * KP + k0 + 8);
      }
      const v16bf bf = (c < 4) ? hB[c] : bx[c - 4];
      #pragma unroll
      for (int i = 0; i < 8; ++i)
        acc[i] = __builtin_amdgcn_wmma_f32_16x16x32_bf16(
                     false, af[i].v, false, bf, (short)0, acc[i], false, false);
    }

    #pragma unroll
    for (int c = 0; c < 4; ++c) {
      unsigned p0[4], p1[4];
      #pragma unroll
      for (int d = 0; d < 4; ++d) {
        p0[d] = pack2(acc[2 * c][2 * d],     acc[2 * c][2 * d + 1]);
        p1[d] = pack2(acc[2 * c + 1][2 * d], acc[2 * c + 1][2 * d + 1]);
      }
      BFrag f;
      #pragma unroll
      for (int d = 0; d < 4; ++d) {
        const unsigned snd = H ? p0[d] : p1[d];
        const unsigned own = H ? p1[d] : p0[d];
        const unsigned rcv = (unsigned)__shfl_xor((int)snd, 16, 32);
        f.u[d]     = H ? rcv : own;
        f.u[4 + d] = H ? own : rcv;
      }
      hB[c] = f.v;
    }
  }

  #pragma unroll
  for (int i = 0; i < 8; ++i) {
    v8f o;
    #pragma unroll
    for (int v = 0; v < 8; ++v) o[v] = tanhf(acc[i][v]);
    *(v8f*)(out + (size_t)row * HID + i * 16 + 8 * H) = o;
  }
}

extern "C" void kernel_launch(void* const* d_in, const int* in_sizes, int n_in,
                              void* d_out, int out_size, void* d_ws, size_t ws_size,
                              hipStream_t stream) {
  (void)n_in; (void)out_size;
  const int*   xpad = (const int*)d_in[0];    // [B, 51] int32
  const float* emb  = (const float*)d_in[1];  // [50000, 64]
  const float* whx  = (const float*)d_in[2];  // [51, 64, 128]
  const float* whh  = (const float*)d_in[3];  // [51, 128, 128]
  const float* bh   = (const float*)d_in[4];  // [51, 1, 128]
  float* out = (float*)d_out;                 // [B, 128]

  const int B = in_sizes[0] / T_STEPS;        // 8192
  const size_t need = (size_t)T_STEPS * HID * KW * sizeof(unsigned short);  // ~2.4 MB

  if (ws_size >= need) {
    // one-time bf16 transpose-convert of all step weights into workspace
    const long pairs = (long)T_STEPS * HID * (KW / 2);
    hipLaunchKernelGGL(convert_weights_kernel, dim3((unsigned)(pairs / 256)), dim3(256),
                       0, stream, whx, whh, (unsigned*)d_ws);
    hipLaunchKernelGGL(rnn_wmma_tdm_kernel, dim3(B / 64), dim3(128), 0, stream,
                       xpad, emb, bh, (const unsigned short*)d_ws, out);
  } else {
    hipLaunchKernelGGL(rnn_wmma_stage_kernel, dim3(B / 64), dim3(128), 0, stream,
                       xpad, emb, whx, whh, bh, out);
  }
}
